// IndConv2_27788438405782
// MI455X (gfx1250) — matrix operand order repaired and checked by reading, not compiled, not run on hardware
//
#include <hip/hip_runtime.h>

typedef __attribute__((ext_vector_type(8)))  float  v8f;
typedef __attribute__((ext_vector_type(4)))  __bf16 v4bf;
typedef __attribute__((ext_vector_type(8)))  __bf16 v8bf;
typedef __attribute__((ext_vector_type(16))) __bf16 v16bf;

#define BATCH   16384
#define CCH     128
#define OCH     128
#define K_DIM   1152   // C * 9
#define N_DIM   1152   // O * 9
#define KT      32     // K per LDS stage == K per bf16 WMMA
#define STR     40     // bf16 elems per LDS row (32 + 8 pad) -> 80B rows, 16B aligned

// ---------------------------------------------------------------------------
// Prep: fold nearest-upsample multiplicities into weights (f32 in workspace).
// WeffT[n][k], n = o*9 + p  (GEMM output row == final (B,O,3,3) layout)
//             k = c*9 + ii*3 + jj (matches x's (B,C,3,3) flat layout)
// ---------------------------------------------------------------------------
__global__ __launch_bounds__(256) void IndConv2_prep(
    const float* __restrict__ W,     // (9, O, C, 3, 3)
    const float* __restrict__ bias,  // (9, O)
    float* __restrict__ WeffT,       // (N_DIM, K_DIM)
    float* __restrict__ biasN)       // (N_DIM)
{
    const int idx = blockIdx.x * 256 + threadIdx.x;
    if (idx < N_DIM) {
        const int o = idx / 9, p = idx % 9;
        biasN[idx] = bias[p * OCH + o];
    }
    if (idx >= N_DIM * K_DIM) return;

    const int n = idx / K_DIM, k = idx % K_DIM;
    const int o = n / 9, p = n % 9;
    const int r = p / 3, cpos = p % 3;
    const int c = k / 9, rem = k % 9;
    const int ii = rem / 3, jj = rem % 3;

    const float* Wp = W + ((size_t)(p * OCH + o) * CCH + c) * 9;
    float acc = 0.0f;
#pragma unroll
    for (int i = 0; i < 3; ++i) {
        if ((2 + r + i) / 3 != ii) continue;
#pragma unroll
        for (int j = 0; j < 3; ++j) {
            if ((2 + cpos + j) / 3 == jj) acc += Wp[i * 3 + j];
        }
    }
    WeffT[(size_t)n * K_DIM + k] = acc;
}

// ---------------------------------------------------------------------------
// helpers
// ---------------------------------------------------------------------------
__device__ __forceinline__ void split4(const float4 f, v4bf& h, v4bf& l) {
    const float c[4] = {f.x, f.y, f.z, f.w};
#pragma unroll
    for (int i = 0; i < 4; ++i) {
        __bf16 hi = (__bf16)c[i];
        h[i] = hi;
        l[i] = (__bf16)(c[i] - (float)hi);
    }
}

// A 16x32 bf16 fragment: elements 0-7 = K[8*laneHi..), 8-15 = K[16+8*laneHi..)
__device__ __forceinline__ v16bf frag_a(const __bf16* buf, int rowBase,
                                        int lane15, int laneHi) {
    const __bf16* p = buf + (rowBase + lane15) * STR;
    v8bf a0 = *(const v8bf*)(p + laneHi * 8);
    v8bf a1 = *(const v8bf*)(p + 16 + laneHi * 8);
    return __builtin_shufflevector(a0, a1, 0, 1, 2, 3, 4, 5, 6, 7,
                                   8, 9, 10, 11, 12, 13, 14, 15);
}

// B 32x16 bf16 fragment: elements 0-7 = K[8*laneHi..), 8-15 = K[16+8*laneHi..)
__device__ __forceinline__ v16bf frag_b(const __bf16* buf, int rowBase,
                                        int lane15, int laneHi) {
    const __bf16* p = buf + (rowBase + lane15) * STR;
    v8bf b0 = *(const v8bf*)(p + laneHi * 8);
    v8bf b1 = *(const v8bf*)(p + 16 + laneHi * 8);
    return __builtin_shufflevector(b0, b1, 0, 1, 2, 3, 4, 5, 6, 7,
                                   8, 9, 10, 11, 12, 13, 14, 15);
}

// ---------------------------------------------------------------------------
// GEMM via bf16x3 split: out[b][n] = sum_k X[b][k]*WT[n][k] + biasN[n]
// Block tile 128x128, 8 waves as 4(M) x 2(N), wave tile 32x64 = 2x4 WMMA tiles.
// ---------------------------------------------------------------------------
__global__ __launch_bounds__(256) void IndConv2_gemm(
    const float* __restrict__ X,      // (BATCH, K_DIM)
    const float* __restrict__ WT,     // (N_DIM, K_DIM)
    const float* __restrict__ biasN,  // (N_DIM)
    float* __restrict__ out)          // (BATCH, N_DIM)
{
    __shared__ __bf16 Ah[128 * STR];
    __shared__ __bf16 Al[128 * STR];
    __shared__ __bf16 Bh[128 * STR];
    __shared__ __bf16 Bl[128 * STR];

    const int nblk  = blockIdx.x % 9;       // M-major grid: X tile reused in L2
    const int mblk  = blockIdx.x / 9;
    const int tileM = mblk * 128;
    const int tileN = nblk * 128;

    const int tid    = threadIdx.x;
    const int lane   = tid & 31;
    const int wid    = tid >> 5;
    const int m_w    = wid >> 1;            // 0..3
    const int n_w    = wid & 1;             // 0..1
    const int lane15 = lane & 15;
    const int laneHi = lane >> 4;           // 0 or 1

    v8f acc[2][4];
#pragma unroll
    for (int mt = 0; mt < 2; ++mt)
#pragma unroll
        for (int nt = 0; nt < 4; ++nt)
            acc[mt][nt] = (v8f)(0.0f);

    // cooperative staging: 128 rows x 32 k (f32) per matrix per stage
    const int col4 = (tid & 7) * 4;         // 0,4,...,28
    const int r0   = tid >> 3;              // 0..31

    for (int k0 = 0; k0 < K_DIM; k0 += KT) {
#pragma unroll
        for (int q = 0; q < 4; ++q) {
            const int row = r0 + q * 32;
            float4 xa = *(const float4*)(X  + (size_t)(tileM + row) * K_DIM + k0 + col4);
            float4 wb = *(const float4*)(WT + (size_t)(tileN + row) * K_DIM + k0 + col4);
            v4bf xh, xl, wh, wl;
            split4(xa, xh, xl);
            split4(wb, wh, wl);
            *(v4bf*)(Ah + row * STR + col4) = xh;
            *(v4bf*)(Al + row * STR + col4) = xl;
            *(v4bf*)(Bh + row * STR + col4) = wh;
            *(v4bf*)(Bl + row * STR + col4) = wl;
        }
        __syncthreads();

        v16bf ah[2], al[2], bh[4], bl[4];
#pragma unroll
        for (int mt = 0; mt < 2; ++mt) {
            const int rb = m_w * 32 + mt * 16;
            ah[mt] = frag_a(Ah, rb, lane15, laneHi);
            al[mt] = frag_a(Al, rb, lane15, laneHi);
        }
#pragma unroll
        for (int nt = 0; nt < 4; ++nt) {
            const int rb = n_w * 64 + nt * 16;
            bh[nt] = frag_b(Bh, rb, lane15, laneHi);
            bl[nt] = frag_b(Bl, rb, lane15, laneHi);
        }

        // hi*hi, then hi*lo, then lo*hi — consecutive WMMAs hit distinct accs
#pragma unroll
        for (int mt = 0; mt < 2; ++mt)
#pragma unroll
            for (int nt = 0; nt < 4; ++nt)
                acc[mt][nt] = __builtin_amdgcn_wmma_f32_16x16x32_bf16(
                    false, ah[mt], false, bh[nt], (short)0, acc[mt][nt], false, false);
#pragma unroll
        for (int mt = 0; mt < 2; ++mt)
#pragma unroll
            for (int nt = 0; nt < 4; ++nt)
                acc[mt][nt] = __builtin_amdgcn_wmma_f32_16x16x32_bf16(
                    false, ah[mt], false, bl[nt], (short)0, acc[mt][nt], false, false);
#pragma unroll
        for (int mt = 0; mt < 2; ++mt)
#pragma unroll
            for (int nt = 0; nt < 4; ++nt)
                acc[mt][nt] = __builtin_amdgcn_wmma_f32_16x16x32_bf16(
                    false, al[mt], false, bh[nt], (short)0, acc[mt][nt], false, false);

        __syncthreads();
    }

    // Epilogue: bias + store. C/D layout: VGPR v -> M = v + 8*laneHi, N = lane15.
#pragma unroll
    for (int nt = 0; nt < 4; ++nt) {
        const int n   = tileN + n_w * 64 + nt * 16 + lane15;
        const float bn = biasN[n];
#pragma unroll
        for (int mt = 0; mt < 2; ++mt) {
            const int rbase = tileM + m_w * 32 + mt * 16 + laneHi * 8;
#pragma unroll
            for (int v = 0; v < 8; ++v)
                out[(size_t)(rbase + v) * N_DIM + n] = acc[mt][nt][v] + bn;
        }
    }
}

// ---------------------------------------------------------------------------
extern "C" void kernel_launch(void* const* d_in, const int* in_sizes, int n_in,
                              void* d_out, int out_size, void* d_ws, size_t ws_size,
                              hipStream_t stream) {
    const float* x  = (const float*)d_in[0];   // (16384, 128, 3, 3)
    const float* W  = (const float*)d_in[1];   // (9, 128, 128, 3, 3)
    const float* b  = (const float*)d_in[2];   // (9, 128)

    float* WeffT = (float*)d_ws;                      // N_DIM * K_DIM floats
    float* biasN = WeffT + (size_t)N_DIM * K_DIM;     // N_DIM floats
    float* out   = (float*)d_out;                     // (16384, 128, 3, 3)

    const int prep_total = N_DIM * K_DIM;
    IndConv2_prep<<<(prep_total + 255) / 256, 256, 0, stream>>>(W, b, WeffT, biasN);

    const int grid = (BATCH / 128) * 9;               // 1152 blocks
    IndConv2_gemm<<<grid, 256, 0, stream>>>(x, WeffT, biasN, out);
}
